// own_1_bigkernels_60962765800152
// MI455X (gfx1250) — compile-verified, matrix operand order checked by reading, not run on hardware
//
#include <hip/hip_runtime.h>

typedef __attribute__((ext_vector_type(16))) __bf16 v16bf;
typedef __attribute__((ext_vector_type(8)))  float  v8f;
typedef __attribute__((ext_vector_type(4)))  unsigned int u32x4;
typedef __attribute__((ext_vector_type(8)))  int i32x8;
typedef __attribute__((ext_vector_type(4)))  int i32x4;

#define NVOX   150000
#define CCH    64
#define KVOFF  27
#define EPSBN  1e-3f
#define NTILES (NVOX / 16)                  // 9375 exactly
#define NPAIRS ((NTILES + 1) / 2)           // 4688 (last pair has one tile)
#define WPK_ELEMS (KVOFF * 2 * 4 * 32 * 16) // 110592 bf16 elems
#define SMEM_BYTES (WPK_ELEMS * 2)          // 221184 B < 320KB WGP LDS
#define SMEM_U64   (SMEM_BYTES / 8)         // 27648 8-byte units (fits 16-bit tile_dim0)
#define STATS_BLOCKS 256

// ---------- helpers ----------
__device__ __forceinline__ unsigned short f2bf_rne(float f) {
    unsigned int u = __float_as_uint(f);
    return (unsigned short)((u + 0x7FFFu + ((u >> 16) & 1u)) >> 16);
}

// ---------- BN stats: per-block partial sums (deterministic) ----------
__global__ void stats_kernel(const float* __restrict__ x, float* __restrict__ partial) {
    __shared__ float sb[256];
    __shared__ float qb[256];
    const int tid = threadIdx.x;
    const int c  = tid & 63;
    const int rg = tid >> 6;
    float s = 0.f, q = 0.f;
    for (int r = blockIdx.x * 4 + rg; r < NVOX; r += gridDim.x * 4) {
        float v = x[(size_t)r * CCH + c];
        s += v; q += v * v;
    }
    sb[tid] = s; qb[tid] = q;
    __syncthreads();
    if (tid < 64) {
        float S = sb[tid] + sb[tid + 64] + sb[tid + 128] + sb[tid + 192];
        float Q = qb[tid] + qb[tid + 64] + qb[tid + 128] + qb[tid + 192];
        partial[(size_t)blockIdx.x * 128 + tid]      = S;
        partial[(size_t)blockIdx.x * 128 + 64 + tid] = Q;
    }
}

// ---------- BN finalize: sequential reduce -> scale/shift ----------
__global__ void finalize_kernel(const float* __restrict__ partial,
                                const float* __restrict__ g, const float* __restrict__ b,
                                float* __restrict__ scsh, int nblk) {
    const int c = threadIdx.x; // 64 threads
    float S = 0.f, Q = 0.f;
    for (int i = 0; i < nblk; ++i) {
        S += partial[(size_t)i * 128 + c];
        Q += partial[(size_t)i * 128 + 64 + c];
    }
    float mean = S / (float)NVOX;
    float var  = Q / (float)NVOX - mean * mean;
    float inv  = rsqrtf(var + EPSBN);
    float sc   = inv * g[c];
    scsh[c]      = sc;
    scsh[64 + c] = b[c] - mean * sc;
}

// ---------- BN apply + ReLU + fp32->bf16, write padded (N+1)x64 ----------
__global__ void apply_kernel(const float* __restrict__ x, const float* __restrict__ scsh,
                             unsigned short* __restrict__ ybf) {
    const int total = (NVOX + 1) * CCH;
    for (int i = blockIdx.x * blockDim.x + threadIdx.x; i < total;
         i += gridDim.x * blockDim.x) {
        if (i < NVOX * CCH) {
            int c = i & 63;
            float v = fmaxf(fmaf(x[i], scsh[c], scsh[64 + c]), 0.f);
            ybf[i] = f2bf_rne(v);
        } else {
            ybf[i] = 0;  // zero pad row (gathers of absent neighbors land here)
        }
    }
}

// ---------- pack fp32 weights (27,64,64) into bf16 WMMA B-fragment layout ----------
// packed index p = (((k*2 + ks)*4 + nt)*32 + lane)*16 + h
__global__ void pack_kernel(const float* __restrict__ W, unsigned short* __restrict__ wpk) {
    int p = blockIdx.x * blockDim.x + threadIdx.x;
    if (p >= WPK_ELEMS) return;
    int h    = p & 15;
    int lane = (p >> 4) & 31;
    int nt   = (p >> 9) & 3;
    int ks   = (p >> 11) & 1;
    int k    = p >> 12;
    int c0   = (lane >= 16) ? 8 : 0;
    int cin  = ks * 32 + ((h < 8) ? (c0 + h) : (16 + c0 + (h - 8)));
    int cout = nt * 16 + (lane & 15);
    float v = W[(size_t)k * 4096 + (size_t)cin * 64 + cout];
    wpk[p] = f2bf_rne(v);
}

// ---------- gather-GEMM conv: one wave per PAIR of 16-row tiles, weights via TDM->LDS ----------
__global__ __launch_bounds__(512, 1) void conv_kernel(
    const unsigned short* __restrict__ ybf,   // (N+1) x 64 bf16
    const int* __restrict__ nb,               // 27 x N
    const unsigned short* __restrict__ wpk,   // packed weights (WPK_ELEMS)
    const float* __restrict__ resid,          // residual (may be null)
    int has_resid,
    float* __restrict__ out)                  // N x 64 fp32
{
    extern __shared__ uint4 BshQ[];
    unsigned short* Bsh = (unsigned short*)BshQ;

    // ---- Tensor Data Mover: one DMA of all packed weights into LDS (wave 0 only) ----
    if ((threadIdx.x >> 5) == 0) {
        unsigned long long ga = (unsigned long long)wpk;
        unsigned ldsoff = (unsigned)(unsigned long long)Bsh;   // low 32 bits = LDS byte offset
        // D# group0: count=1, lds_addr, global_addr[56:0], type=2
        u32x4 g0 = { 1u, ldsoff, (unsigned)ga,
                     (unsigned)((ga >> 32) & 0x01FFFFFFull) | 0x80000000u };
        // D# group1: data_size=8B; tensor_dim0=27648; tensor_dim1=1; tile_dim0=27648;
        //            tile_dim1=1; dim0_stride=27648; dim1_stride=27648
        i32x8 g1 = { 0x00030000, (int)0x6C000000, 0x00010000, (int)0x6C000000,
                     1, SMEM_U64, (int)0x6C000000, 0 };
        // D# group2: tensor_dim2=1, tensor_dim3=1, dim2_stride=27648, tile_dim3=0
        i32x4 g2 = { 1, 1, SMEM_U64, 0 };
        i32x4 g3 = { 0, 0, 0, 0 };
#if __clang_major__ >= 23
        i32x8 g4 = { 0, 0, 0, 0, 0, 0, 0, 0 };
        __builtin_amdgcn_tensor_load_to_lds(g0, g1, g2, g3, g4, 0);
#else
        __builtin_amdgcn_tensor_load_to_lds(g0, g1, g2, g3, 0);
#endif
        __builtin_amdgcn_s_wait_tensorcnt((short)0);
    }
    __syncthreads();

    const int tid  = threadIdx.x;
    const int lane = tid & 31;
    const int l15  = lane & 15;
    const int hi   = lane >> 4;     // 0 or 1
    const int c0   = hi * 8;        // half-wave K phase
    const int gw   = blockIdx.x * (blockDim.x >> 5) + (tid >> 5);
    const int nw   = gridDim.x * (blockDim.x >> 5);

    union U { uint4 q[2]; v16bf v; };

    for (int p = gw; p < NPAIRS; p += nw) {        // wave-uniform loop
        const int t0 = p * 2;
        const int t1 = t0 + 1;
        const bool has2 = (t1 < NTILES);           // wave-uniform
        const int row0 = t0 * 16 + l15;
        const int row1 = has2 ? (t1 * 16 + l15) : row0;

        v8f a00 = {0.f,0.f,0.f,0.f,0.f,0.f,0.f,0.f};
        v8f a01 = {0.f,0.f,0.f,0.f,0.f,0.f,0.f,0.f};
        v8f a02 = {0.f,0.f,0.f,0.f,0.f,0.f,0.f,0.f};
        v8f a03 = {0.f,0.f,0.f,0.f,0.f,0.f,0.f,0.f};
        v8f a10 = {0.f,0.f,0.f,0.f,0.f,0.f,0.f,0.f};
        v8f a11 = {0.f,0.f,0.f,0.f,0.f,0.f,0.f,0.f};
        v8f a12 = {0.f,0.f,0.f,0.f,0.f,0.f,0.f,0.f};
        v8f a13 = {0.f,0.f,0.f,0.f,0.f,0.f,0.f,0.f};

#pragma unroll 1
        for (int k = 0; k < KVOFF; ++k) {
            const int kb = k * NVOX;
            const int i0 = nb[kb + row0];
            const int i1 = nb[kb + row1];
            const unsigned short* rp0 = ybf + (size_t)i0 * CCH;
            const unsigned short* rp1 = ybf + (size_t)i1 * CCH;
#pragma unroll
            for (int ks = 0; ks < 2; ++ks) {
                U A0, A1;
                A0.q[0] = *(const uint4*)(rp0 + ks * 32 + c0);
                A0.q[1] = *(const uint4*)(rp0 + ks * 32 + 16 + c0);
                A1.q[0] = *(const uint4*)(rp1 + ks * 32 + c0);
                A1.q[1] = *(const uint4*)(rp1 + ks * 32 + 16 + c0);

                const unsigned short* bb = Bsh + (size_t)((k * 2 + ks) * 4) * 512 + lane * 16;
                U B0, B1, B2, B3;   // independent regs -> one ds clause, one wait
                B0.q[0] = *(const uint4*)(bb);        B0.q[1] = *(const uint4*)(bb + 8);
                B1.q[0] = *(const uint4*)(bb + 512);  B1.q[1] = *(const uint4*)(bb + 520);
                B2.q[0] = *(const uint4*)(bb + 1024); B2.q[1] = *(const uint4*)(bb + 1032);
                B3.q[0] = *(const uint4*)(bb + 1536); B3.q[1] = *(const uint4*)(bb + 1544);

                a00 = __builtin_amdgcn_wmma_f32_16x16x32_bf16(false, A0.v, false, B0.v,
                                                              (short)0, a00, false, false);
                a10 = __builtin_amdgcn_wmma_f32_16x16x32_bf16(false, A1.v, false, B0.v,
                                                              (short)0, a10, false, false);
                a01 = __builtin_amdgcn_wmma_f32_16x16x32_bf16(false, A0.v, false, B1.v,
                                                              (short)0, a01, false, false);
                a11 = __builtin_amdgcn_wmma_f32_16x16x32_bf16(false, A1.v, false, B1.v,
                                                              (short)0, a11, false, false);
                a02 = __builtin_amdgcn_wmma_f32_16x16x32_bf16(false, A0.v, false, B2.v,
                                                              (short)0, a02, false, false);
                a12 = __builtin_amdgcn_wmma_f32_16x16x32_bf16(false, A1.v, false, B2.v,
                                                              (short)0, a12, false, false);
                a03 = __builtin_amdgcn_wmma_f32_16x16x32_bf16(false, A0.v, false, B3.v,
                                                              (short)0, a03, false, false);
                a13 = __builtin_amdgcn_wmma_f32_16x16x32_bf16(false, A1.v, false, B3.v,
                                                              (short)0, a13, false, false);
            }
        }

        // C/D layout: lane(0-15): VGPR j = row M=j, col=lane ; lane(16-31): M=8+j, col=lane-16
        {
            const size_t ob = ((size_t)t0 * 16 + (size_t)hi * 8) * 64 + l15;
            if (has_resid) {
#pragma unroll
                for (int j = 0; j < 8; ++j) {
                    size_t o = ob + (size_t)j * 64;
                    out[o]      = a00[j] + resid[o];
                    out[o + 16] = a01[j] + resid[o + 16];
                    out[o + 32] = a02[j] + resid[o + 32];
                    out[o + 48] = a03[j] + resid[o + 48];
                }
            } else {
#pragma unroll
                for (int j = 0; j < 8; ++j) {
                    size_t o = ob + (size_t)j * 64;
                    out[o]      = a00[j];
                    out[o + 16] = a01[j];
                    out[o + 32] = a02[j];
                    out[o + 48] = a03[j];
                }
            }
        }
        if (has2) {
            const size_t ob = ((size_t)t1 * 16 + (size_t)hi * 8) * 64 + l15;
            if (has_resid) {
#pragma unroll
                for (int j = 0; j < 8; ++j) {
                    size_t o = ob + (size_t)j * 64;
                    out[o]      = a10[j] + resid[o];
                    out[o + 16] = a11[j] + resid[o + 16];
                    out[o + 32] = a12[j] + resid[o + 32];
                    out[o + 48] = a13[j] + resid[o + 48];
                }
            } else {
#pragma unroll
                for (int j = 0; j < 8; ++j) {
                    size_t o = ob + (size_t)j * 64;
                    out[o]      = a10[j];
                    out[o + 16] = a11[j];
                    out[o + 32] = a12[j];
                    out[o + 48] = a13[j];
                }
            }
        }
    }
}

// ---------- workspace layout (bytes) ----------
//      0 : scsh   128 f32
//    512 : partial STATS_BLOCKS*128 f32  (131072 B)
// 131584 : wpk1   221184 B
// 352768 : wpk2   221184 B
// 573952 : ybf    (N+1)*64 bf16 = 19200128 B
// 19774080 : z1   N*64 f32 = 38400000 B       total ~58.2 MB

extern "C" void kernel_launch(void* const* d_in, const int* in_sizes, int n_in,
                              void* d_out, int out_size, void* d_ws, size_t ws_size,
                              hipStream_t stream) {
    const float* x  = (const float*)d_in[0];
    const int*   nb = (const int*)  d_in[1];
    const float* W1 = (const float*)d_in[2];
    const float* W2 = (const float*)d_in[3];
    const float* g1 = (const float*)d_in[4];
    const float* b1 = (const float*)d_in[5];
    const float* g2 = (const float*)d_in[6];
    const float* b2 = (const float*)d_in[7];
    float* out = (float*)d_out;

    char* ws = (char*)d_ws;
    float*          scsh    = (float*)(ws + 0);
    float*          partial = (float*)(ws + 512);
    unsigned short* wpk1    = (unsigned short*)(ws + 131584);
    unsigned short* wpk2    = (unsigned short*)(ws + 352768);
    unsigned short* ybf     = (unsigned short*)(ws + 573952);
    float*          z1      = (float*)(ws + 19774080);

    // weight packing (independent of BN pipeline)
    pack_kernel<<<(WPK_ELEMS + 255) / 256, 256, 0, stream>>>(W1, wpk1);
    pack_kernel<<<(WPK_ELEMS + 255) / 256, 256, 0, stream>>>(W2, wpk2);

    // stage 1: BNReLU(x) -> ybf ; conv1 -> z1
    stats_kernel   <<<STATS_BLOCKS, 256, 0, stream>>>(x, partial);
    finalize_kernel<<<1, 64, 0, stream>>>(partial, g1, b1, scsh, STATS_BLOCKS);
    apply_kernel   <<<2048, 256, 0, stream>>>(x, scsh, ybf);
    conv_kernel    <<<128, 512, SMEM_BYTES, stream>>>(ybf, nb, wpk1, nullptr, 0, z1);

    // stage 2: BNReLU(z1) -> ybf ; conv2 + residual -> out
    stats_kernel   <<<STATS_BLOCKS, 256, 0, stream>>>(z1, partial);
    finalize_kernel<<<1, 64, 0, stream>>>(partial, g2, b2, scsh, STATS_BLOCKS);
    apply_kernel   <<<2048, 256, 0, stream>>>(z1, scsh, ybf);
    conv_kernel    <<<128, 512, SMEM_BYTES, stream>>>(ybf, nb, wpk2, x, 1, out);
}
